// Attention_layer_attention_logits_75222057222599
// MI455X (gfx1250) — matrix operand order repaired and checked by reading, not compiled, not run on hardware
//
#include <hip/hip_runtime.h>

typedef __attribute__((ext_vector_type(16))) __bf16 v16bf;
typedef __attribute__((ext_vector_type(8)))  float  v8f;
typedef __attribute__((ext_vector_type(4)))  unsigned int v4u;
typedef __attribute__((ext_vector_type(8)))  int    v8i;
typedef __attribute__((ext_vector_type(4)))  int    v4i;

constexpr int D  = 512;
constexpr int Bb = 64;
constexpr int Cc = 8;
constexpr int LQ = 128;
constexpr int LK = 128;

// ---------------- WMMA fragment loaders (wave32, 16x16x32 bf16) ----------------
// A 16x32 (MxK): lane m=lane&15; lane<16: elems = K0..15 ; lane>=16: K16..31
static __device__ inline v16bf a_frag_f32(const float* p0, int ld) {
  int lane = threadIdx.x & 31;
  int m = lane & 15;
  int kb = (lane & 16) ? 16 : 0;
  const float* p = p0 + (size_t)m * ld + kb;
  v16bf a;
#pragma unroll
  for (int e = 0; e < 8; ++e) { a[e] = (__bf16)p[e]; a[e + 8] = (__bf16)p[e + 8]; }
  return a;
}
static __device__ inline v16bf a_frag_bf16(const __bf16* p0, int ld) {
  int lane = threadIdx.x & 31;
  int m = lane & 15;
  int kb = (lane & 16) ? 16 : 0;
  const __bf16* p = p0 + (size_t)m * ld + kb;
  v16bf a;
#pragma unroll
  for (int e = 0; e < 8; ++e) { a[e] = p[e]; a[e + 8] = p[e + 8]; }
  return a;
}
// A where memory holds S[k][m] (A = S^T): strided
static __device__ inline v16bf aT_frag_f32(const float* p0, int ld) {
  int lane = threadIdx.x & 31;
  int m = lane & 15;
  int kb = (lane & 16) ? 16 : 0;
  v16bf a;
#pragma unroll
  for (int e = 0; e < 8; ++e) {
    a[e]     = (__bf16)p0[(size_t)(kb + e) * ld + m];
    a[e + 8] = (__bf16)p0[(size_t)(kb + 8 + e) * ld + m];
  }
  return a;
}
// B 32x16 (KxN) where memory holds S[n][k] (B = S^T): contiguous per lane.
// lane n=lane&15; lane<16 holds K0..15, lane>=16 holds K16..31
static __device__ inline v16bf bT_frag_f32(const float* p0, int ld) {
  int lane = threadIdx.x & 31;
  int n = lane & 15;
  int kb = (lane & 16) ? 16 : 0;
  const float* p = p0 + (size_t)n * ld + kb;
  v16bf b;
#pragma unroll
  for (int e = 0; e < 16; ++e) b[e] = (__bf16)p[e];
  return b;
}
static __device__ inline v16bf bT_frag_bf16(const __bf16* p0, int ld) {
  int lane = threadIdx.x & 31;
  int n = lane & 15;
  int kb = (lane & 16) ? 16 : 0;
  const __bf16* p = p0 + (size_t)n * ld + kb;
  v16bf b;
#pragma unroll
  for (int e = 0; e < 16; ++e) b[e] = p[e];
  return b;
}
// B from row-major src[k][n]: strided
static __device__ inline v16bf b_frag_f32(const float* p0, int ld) {
  int lane = threadIdx.x & 31;
  int n = lane & 15;
  int kb = (lane & 16) ? 16 : 0;
  v16bf b;
#pragma unroll
  for (int e = 0; e < 16; ++e) b[e] = (__bf16)p0[(size_t)(kb + e) * ld + n];
  return b;
}
// D/C 16x16 f32: vgpr r -> row m=r+(lane>=16?8:0), col n=lane&15
static __device__ inline void store_tile_bf16(__bf16* p0, int ld, const v8f& c) {
  int lane = threadIdx.x & 31;
  int n = lane & 15;
  int mb = (lane & 16) ? 8 : 0;
#pragma unroll
  for (int r = 0; r < 8; ++r) p0[(size_t)(mb + r) * ld + n] = (__bf16)c[r];
}
static __device__ inline void store_tile_f32(float* p0, int ld, const v8f& c) {
  int lane = threadIdx.x & 31;
  int n = lane & 15;
  int mb = (lane & 16) ? 8 : 0;
#pragma unroll
  for (int r = 0; r < 8; ++r) p0[(size_t)(mb + r) * ld + n] = c[r];
}

#define WMMA_BF16(a, b, c) \
  __builtin_amdgcn_wmma_f32_16x16x32_bf16(false, (a), false, (b), (short)0, (c), false, false)

// ---------------- TDM helpers ----------------------------------------------------------
typedef __attribute__((address_space(3))) void* lds_vptr_t;
static __device__ inline unsigned lds_offset_of(void* p) {
  return (unsigned)(unsigned long long)(lds_vptr_t)p;
}

// DMA a [128 rows(l) x 32 cols(d)] f32 tile (row stride 512 f32) from global into LDS,
// padding each 32-dword row with 4 dwords -> LDS row stride 36 dwords (bank-conflict-free).
static __device__ inline void tdm_load_key_chunk(const float* gsrc, unsigned lds_bytes) {
  unsigned long long ga = (unsigned long long)gsrc;
  v4u g0;
  g0[0] = 1u;                                           // count=1, user descriptor
  g0[1] = lds_bytes;                                    // lds_addr
  g0[2] = (unsigned)(ga & 0xffffffffu);                 // global_addr[31:0]
  g0[3] = (unsigned)((ga >> 32) & 0x01ffffffu) | (2u << 30);  // addr[56:32] | type=2
  v8i g1;
  g1[0] = (2 << 16) | (1 << 20) | (4 << 22) | (3 << 25);  // data_size=4B, pad_en,
                                                          // pad_interval=32dw, pad_amount=4dw
  g1[1] = (int)(512u << 16);   // tensor_dim0 = 512 (d)
  g1[2] = (int)(128u << 16);   // tensor_dim1 = 128 (l)
  g1[3] = (int)(32u << 16);    // tile_dim0 = 32
  g1[4] = 128;                 // tile_dim1 = 128, tile_dim2 = 0
  g1[5] = 512;                 // tensor_dim0_stride = 512 (low 32)
  g1[6] = 0;
  g1[7] = 0;
  v4i z = {0, 0, 0, 0};
#if defined(__clang_major__) && (__clang_major__ >= 23)
  v8i z8 = {0, 0, 0, 0, 0, 0, 0, 0};
  __builtin_amdgcn_tensor_load_to_lds(g0, g1, z, z, z8, 0);
#else
  __builtin_amdgcn_tensor_load_to_lds(g0, g1, z, z, 0);
#endif
}

// ---------------- Kernel 1: Mt[d2][d] = sum_e wk[e][d2] * wq[e][d]  (= (wq^T wk)^T) ----
__global__ __launch_bounds__(128) void k_wqwk(const float* __restrict__ wq,
                                              const float* __restrict__ wk,
                                              __bf16* __restrict__ Mt) {
  int tile = blockIdx.x * 4 + (threadIdx.x >> 5);  // 1024 tiles over [512,512]
  int tm = tile >> 5, tn = tile & 31;
  v8f acc = {};
  for (int kk = 0; kk < D; kk += 32) {
    v16bf a = aT_frag_f32(wk + (size_t)kk * D + tm * 16, D);  // A[m=d2][k=e]
    v16bf b = b_frag_f32(wq + (size_t)kk * D + tn * 16, D);   // B[k=e][n=d]
    acc = WMMA_BF16(a, b, acc);
  }
  store_tile_bf16(Mt + (size_t)(tm * 16) * D + tn * 16, D, acc);
}

// ---------------- Kernel 2: qm[q][d2] = scale * sum_d query[q][d] * Mt[d2][d] ----------
__global__ __launch_bounds__(128) void k_qproj(const float* __restrict__ query,
                                               const __bf16* __restrict__ Mt,
                                               __bf16* __restrict__ qm) {
  int tile = blockIdx.x * 4 + (threadIdx.x >> 5);  // 16384 tiles over [8192,512]
  int tm = tile >> 5, tn = tile & 31;
  v8f acc = {};
  for (int kk = 0; kk < D; kk += 32) {
    v16bf a = a_frag_f32(query + (size_t)(tm * 16) * D + kk, D);
    v16bf b = bT_frag_bf16(Mt + (size_t)(tn * 16) * D + kk, D);  // B[k=d][n=d2]=Mt[n][k]
    acc = WMMA_BF16(a, b, acc);
  }
  const float scale = 0.04419417382415922f;  // 1/sqrt(512)
#pragma unroll
  for (int r = 0; r < 8; ++r) acc[r] *= scale;
  store_tile_bf16(qm + (size_t)(tm * 16) * D + tn * 16, D, acc);
}

// ---------------- Kernel 3: per (b,c): TDM-staged scores GEMM + masked col-max + ksum --
__global__ __launch_bounds__(256) void k_scores(const __bf16* __restrict__ qm,
                                                const float* __restrict__ key,
                                                const int* __restrict__ mask,
                                                float* __restrict__ sv,
                                                float* __restrict__ ksum) {
  constexpr int KCH = 32;       // D-chunk per WMMA step
  constexpr int NCH = D / KCH;  // 16
  constexpr int LDW = 36;       // padded LDS row stride in dwords
  __shared__ float sbuf[2][LK][LDW];   // double-buffered key chunk, TDM-padded rows
  __shared__ float svpart[8][LK];

  int bc = blockIdx.x;  // 0..511
  int c = bc & (Cc - 1);
  int b = bc >> 3;
  int wave = threadIdx.x >> 5;
  int lane = threadIdx.x & 31;
  int tid = threadIdx.x;

  const __bf16* Abase = qm + ((size_t)b * LQ + (size_t)wave * 16) * D;
  const float*  Kbase = key + (size_t)bc * LK * D;

  if (wave == 0) tdm_load_key_chunk(Kbase, lds_offset_of(&sbuf[0][0][0]));

  v8f acc[8] = {};
  for (int i = 0; i < NCH; ++i) {
    __syncthreads();  // everyone done reading the buffer that the next DMA overwrites
    if (wave == 0) {
      if (i + 1 < NCH) {
        tdm_load_key_chunk(Kbase + (size_t)(i + 1) * KCH,
                           lds_offset_of(&sbuf[(i + 1) & 1][0][0]));
        __builtin_amdgcn_s_wait_tensorcnt(1);  // chunk i landed; chunk i+1 in flight
      } else {
        __builtin_amdgcn_s_wait_tensorcnt(0);
      }
    }
    __syncthreads();  // chunk i visible to all waves
    const float* Bc = &sbuf[i & 1][0][0];
    v16bf a = a_frag_bf16(Abase + (size_t)i * KCH, D);
#pragma unroll
    for (int t = 0; t < 8; ++t) {
      v16bf bt = bT_frag_f32(Bc + (size_t)(t * 16) * LDW, LDW);  // B[k=d][n=l]=key[l][d]
      acc[t] = WMMA_BF16(a, bt, acc[t]);
    }
  }

  // column-max over the wave's 16 q-rows, then across waves via LDS
#pragma unroll
  for (int t = 0; t < 8; ++t) {
    float mx = acc[t][0];
#pragma unroll
    for (int r = 1; r < 8; ++r) mx = fmaxf(mx, acc[t][r]);
    mx = fmaxf(mx, __shfl_xor(mx, 16, 32));  // merge the two row-halves
    if (lane < 16) svpart[wave][t * 16 + lane] = mx;
  }
  __syncthreads();
  if (tid < LK) {
    float m = svpart[0][tid];
#pragma unroll
    for (int w = 1; w < 8; ++w) m = fmaxf(m, svpart[w][tid]);
    if (mask[c * LK + tid] != 0) m = -1e9f;
    sv[(size_t)bc * LK + tid] = m;
  }

  // ksum[d] = sum_l key[l][d]  (key tile is hot in cache from the DMA stream above)
  for (int d = tid; d < D; d += 256) {
    float s = 0.f;
    for (int l = 0; l < LK; ++l) s += Kbase[(size_t)l * D + d];
    ksum[(size_t)bc * D + d] = s;
  }
}

// ---------------- Kernel 4: vsum[bc][e] = sum_d ksum[bc][d] * wv[e][d] ------------------
__global__ __launch_bounds__(128) void k_vsum(const float* __restrict__ ksum,
                                              const float* __restrict__ wv,
                                              float* __restrict__ vsum) {
  int tile = blockIdx.x * 4 + (threadIdx.x >> 5);  // 1024 tiles over [512,512]
  int tm = tile >> 5, tn = tile & 31;
  v8f acc = {};
  for (int kk = 0; kk < D; kk += 32) {
    v16bf a = a_frag_f32(ksum + (size_t)(tm * 16) * D + kk, D);
    v16bf b = bT_frag_f32(wv + (size_t)(tn * 16) * D + kk, D);  // B[k=d][n=e]=wv[n][k]
    acc = WMMA_BF16(a, b, acc);
  }
  store_tile_f32(vsum + (size_t)(tm * 16) * D + tn * 16, D, acc);
}

// ---------------- Kernel 5: LN epilogue (rank-1 trick) ---------------------------------
// out[bc][l][e] = sv*(vsum[e]-mu)*rsqrt(sv^2*var+eps)*gamma[e]+beta[e]
__global__ __launch_bounds__(256) void k_epilogue(const float* __restrict__ vsum,
                                                  const float* __restrict__ sv,
                                                  const float* __restrict__ gamma,
                                                  const float* __restrict__ beta,
                                                  float* __restrict__ out) {
  int bc = blockIdx.x;
  int lc = blockIdx.y;   // 4 chunks of 32 l-values
  int tid = threadIdx.x; // 256

  float x0 = vsum[(size_t)bc * D + tid];
  float x1 = vsum[(size_t)bc * D + tid + 256];
  float s = x0 + x1, ss = x0 * x0 + x1 * x1;
#pragma unroll
  for (int off = 16; off > 0; off >>= 1) {
    s  += __shfl_down(s, off, 32);
    ss += __shfl_down(ss, off, 32);
  }
  __shared__ float rs[8], rss[8], mv[2];
  if ((tid & 31) == 0) { rs[tid >> 5] = s; rss[tid >> 5] = ss; }
  __syncthreads();
  if (tid == 0) {
    float S = 0.f, SS = 0.f;
#pragma unroll
    for (int i = 0; i < 8; ++i) { S += rs[i]; SS += rss[i]; }
    float mean = S * (1.0f / D);
    mv[0] = mean;
    mv[1] = SS * (1.0f / D) - mean * mean;
  }
  __syncthreads();
  float mean = mv[0], var = mv[1];
  float v0 = x0 - mean, v1 = x1 - mean;
  float g0 = gamma[tid], g1 = gamma[tid + 256];
  float b0 = beta[tid],  b1 = beta[tid + 256];

  for (int li = 0; li < 32; ++li) {
    int l = lc * 32 + li;
    float svl = sv[(size_t)bc * LK + l];
    float a = svl * rsqrtf(svl * svl * var + 1e-5f);
    size_t base = ((size_t)bc * LK + l) * D;
    out[base + tid]       = a * v0 * g0 + b0;
    out[base + tid + 256] = a * v1 * g1 + b1;
  }
}

// ---------------- Host launcher --------------------------------------------------------
extern "C" void kernel_launch(void* const* d_in, const int* in_sizes, int n_in,
                              void* d_out, int out_size, void* d_ws, size_t ws_size,
                              hipStream_t stream) {
  (void)in_sizes; (void)n_in; (void)out_size; (void)ws_size;
  const float* query = (const float*)d_in[0];
  const float* key   = (const float*)d_in[1];
  const int*   mask  = (const int*)d_in[2];
  const float* wq    = (const float*)d_in[3];
  const float* wk    = (const float*)d_in[4];
  const float* wv    = (const float*)d_in[5];
  const float* gamma = (const float*)d_in[6];
  const float* beta  = (const float*)d_in[7];
  float* out = (float*)d_out;

  char* ws = (char*)d_ws;
  size_t off = 0;
  __bf16* Mt  = (__bf16*)(ws + off); off += (size_t)D * D * 2;            // 512 KB
  __bf16* qm  = (__bf16*)(ws + off); off += (size_t)Bb * LQ * D * 2;      // 8 MB
  float*  sv  = (float*)(ws + off);  off += (size_t)Bb * Cc * LK * 4;     // 256 KB
  float*  ksum= (float*)(ws + off);  off += (size_t)Bb * Cc * D * 4;      // 1 MB
  float*  vsum= (float*)(ws + off);                                       // 1 MB

  k_wqwk   <<<256,  128, 0, stream>>>(wq, wk, Mt);
  k_qproj  <<<4096, 128, 0, stream>>>(query, Mt, qm);
  k_scores <<<512,  256, 0, stream>>>(qm, key, mask, sv, ksum);
  k_vsum   <<<256,  128, 0, stream>>>(ksum, wv, vsum);
  dim3 g5(512, 4);
  k_epilogue<<<g5, 256, 0, stream>>>(vsum, sv, gamma, beta, out);
}
